// ViT_85023172592607
// MI455X (gfx1250) — matrix operand order repaired and checked by reading, not compiled, not run on hardware
//
#include <hip/hip_runtime.h>
#include <hip/hip_bf16.h>
#include <cmath>

// ---------------------------------------------------------------------------
// ViT-Base forward for MI455X (gfx1250, wave32, WMMA 16x16x32 bf16)
// GEMMs: LDS double-buffered tiles staged with GLOBAL_LOAD_ASYNC_TO_LDS_B128
// (ASYNCcnt / s_wait_asynccnt), fragments read via ds_load_b128, math via
// v_wmma_f32_16x16x32_bf16.
// ---------------------------------------------------------------------------

typedef __attribute__((ext_vector_type(16))) __bf16 v16bf;
typedef __attribute__((ext_vector_type(8)))  float  v8f;
typedef int v4i __attribute__((vector_size(16)));   // matches builtin's b128 type

#define DEVINL __device__ __forceinline__

// Model constants
static const int NB   = 32;      // batch
static const int NT   = 197;     // tokens (196 patches + cls)
static const int ND   = 768;     // model dim
static const int NH   = 12;      // heads
static const int NHD  = 64;      // head dim
static const int NPAT = 196;     // patches per image
static const int NL   = 12;      // layers
static const int NC   = 1000;    // classes
static const int TOK  = NB * NT;   // 6304  (divisible by 32)
static const int PROW = NB * NPAT; // 6272  (divisible by 32)
static const int SPAD = 224;       // padded score row (f32 elems), 224 = 7*32
static const int NBH  = NB * NH;   // 384

// LDS tile geometry: block tile 128(M) x 64(N), K-step 32.
static const int LROW = 40;              // padded row slot (elements) -> 80 B
static const int LDS_A = 128 * LROW;     // A slab (elements)
static const int LDS_B = 64 * LROW;      // B slab (elements)
static const int LDS_TILE = LDS_A + LDS_B;

// GEMM epilogue flags
enum { GF_BIAS = 1, GF_RESID = 2, GF_GELU = 4, GF_F32 = 8, GF_BF16 = 16 };

// ---------------------------------------------------------------------------
// Async global->LDS staging (CDNA5 path), with synchronous fallback so the
// file compiles on toolchains lacking the builtin.
// ---------------------------------------------------------------------------
#if defined(__has_builtin)
#if __has_builtin(__builtin_amdgcn_global_load_async_to_lds_b128)
#define HAVE_ASYNC_LDS 1
#endif
#if __has_builtin(__builtin_amdgcn_s_wait_asynccnt)
#define HAVE_WAIT_ASYNC 1
#endif
#endif

DEVINL void stage16(const __hip_bfloat16* g, __hip_bfloat16* l) {
#if defined(HAVE_ASYNC_LDS)
  __builtin_amdgcn_global_load_async_to_lds_b128(
      (__attribute__((address_space(1))) v4i*)(g),
      (__attribute__((address_space(3))) v4i*)(l), 0, 0);
#else
  *reinterpret_cast<uint4*>(l) = *reinterpret_cast<const uint4*>(g);
#endif
}

DEVINL void wait_async_all() {
#if defined(HAVE_ASYNC_LDS)
#if defined(HAVE_WAIT_ASYNC)
  __builtin_amdgcn_s_wait_asynccnt(0);
#else
  asm volatile("s_wait_asynccnt 0" ::: "memory");
#endif
#endif
}

// ---------------------------------------------------------------------------
// LDS fragment loaders (bf16, 16 rows x 32 K, row slot = LROW elements)
// A-matrix layout (ISA 7.12.2, 16-bit A 16x32): lane L holds row M=L&15;
//   VGPR0-3: K = (L>>4)*8 .. +7 ; VGPR4-7: K = 16 + (L>>4)*8 .. +7
// B-matrix layout (per sparse-B doc): lane L holds col N=L&15;
//   VGPR0-3: K = (L>>4)*8 .. +7 ; VGPR4-7: K = 16 + (L>>4)*8 .. +7 (same as A)
// ---------------------------------------------------------------------------
DEVINL v16bf lds_fragA(const __hip_bfloat16* base, int row0, int lane) {
  const __hip_bfloat16* p = base + (row0 + (lane & 15)) * LROW + ((lane >> 4) * 8);
  union { uint4 u[2]; v16bf v; } f;
  f.u[0] = *reinterpret_cast<const uint4*>(p);
  f.u[1] = *reinterpret_cast<const uint4*>(p + 16);
  return f.v;
}

DEVINL v16bf lds_fragB(const __hip_bfloat16* base, int row0, int lane) {
  const __hip_bfloat16* p = base + (row0 + (lane & 15)) * LROW + ((lane >> 4) * 8);
  union { uint4 u[2]; v16bf v; } f;
  f.u[0] = *reinterpret_cast<const uint4*>(p);
  f.u[1] = *reinterpret_cast<const uint4*>(p + 16);
  return f.v;
}

// ---------------------------------------------------------------------------
// Generic batched GEMM:  C[z] = A[z] (MxK, bf16) * W[z]^T (W is [N,K] bf16)
// Batch index z decomposed as (zb, zh) = (z / Hdim, z % Hdim); per-operand
// offsets are zb*s?B + zh*s?H.  Block = 8 waves, tile 128x64; wave = 32x32
// via 2x2 WMMA accumulators.  K staged through double-buffered LDS.
// ---------------------------------------------------------------------------
template <int FLAGS>
__global__ __launch_bounds__(256)
void gemm_wmma(const __hip_bfloat16* __restrict__ A, int lda,
               const __hip_bfloat16* __restrict__ W, int ldw,
               float* __restrict__ Cf, __hip_bfloat16* __restrict__ Cb, int ldc,
               const float* __restrict__ bias,
               const float* __restrict__ resid, int ldr,
               int M, int N, int K,
               long long saB, long long saH,
               long long swB, long long swH,
               long long scB, long long scH, int Hdim) {
  __shared__ __align__(16) __hip_bfloat16 lds[2][LDS_TILE];

  const int tid  = threadIdx.x;
  const int lane = tid & 31;
  const int wave = tid >> 5;
  const int mBlk = blockIdx.y * 128;                 // block M origin
  const int nBlk = blockIdx.x * 64;                  // block N origin
  const int mLoc = (wave & 3) * 32;                  // wave M offset in tile
  const int nLoc = (wave >> 2) * 32;                 // wave N offset in tile

  const int zb = (int)blockIdx.z / Hdim;
  const int zh = (int)blockIdx.z % Hdim;
  A += zb * saB + zh * saH;
  W += zb * swB + zh * swH;
  const long long coff = zb * scB + zh * scH;

  // Cooperative stage of one K-slab (A: 128x32, B: 64x32) into lds[buf].
  // 512 + 256 chunks of 16 B; every thread issues 3 async b128 transfers.
  auto stage_tile = [&](int k, int buf) {
#pragma unroll
    for (int c = tid; c < 512; c += 256) {
      int r = c >> 2, seg = c & 3;
      int gr = mBlk + r; if (gr > M - 1) gr = M - 1;
      stage16(A + (long long)gr * lda + k + seg * 8, &lds[buf][r * LROW + seg * 8]);
    }
    {
      int c = tid;            // exactly one chunk per thread
      int r = c >> 2, seg = c & 3;
      int gr = nBlk + r; if (gr > N - 1) gr = N - 1;
      stage16(W + (long long)gr * ldw + k + seg * 8, &lds[buf][LDS_A + r * LROW + seg * 8]);
    }
  };

  const v8f vzero = {0.f, 0.f, 0.f, 0.f, 0.f, 0.f, 0.f, 0.f};
  v8f c00 = vzero, c01 = vzero, c10 = vzero, c11 = vzero;

  int buf = 0;
  stage_tile(0, 0);
  for (int k = 0; k < K; k += 32) {
    wait_async_all();
    __syncthreads();                       // lds[buf] ready for all waves
    if (k + 32 < K) stage_tile(k + 32, buf ^ 1);   // prefetch next slab
    v16bf a0 = lds_fragA(&lds[buf][0],     mLoc,      lane);
    v16bf a1 = lds_fragA(&lds[buf][0],     mLoc + 16, lane);
    v16bf b0 = lds_fragB(&lds[buf][LDS_A], nLoc,      lane);
    v16bf b1 = lds_fragB(&lds[buf][LDS_A], nLoc + 16, lane);
    c00 = __builtin_amdgcn_wmma_f32_16x16x32_bf16(false, a0, false, b0, (short)0, c00, false, false);
    c01 = __builtin_amdgcn_wmma_f32_16x16x32_bf16(false, a0, false, b1, (short)0, c01, false, false);
    c10 = __builtin_amdgcn_wmma_f32_16x16x32_bf16(false, a1, false, b0, (short)0, c10, false, false);
    c11 = __builtin_amdgcn_wmma_f32_16x16x32_bf16(false, a1, false, b1, (short)0, c11, false, false);
    buf ^= 1;
  }

  // Epilogue. C/D layout: VGPR v -> M = v + (lane>=16)*8, N = lane&15.
  const int rowBase = (lane >> 4) * 8;
  const int colBase = lane & 15;
  const int m0 = mBlk + mLoc;
  const int n0 = nBlk + nLoc;
  v8f acc[2][2] = {{c00, c01}, {c10, c11}};
#pragma unroll
  for (int i = 0; i < 2; ++i)
#pragma unroll
    for (int j = 0; j < 2; ++j)
#pragma unroll
      for (int v = 0; v < 8; ++v) {
        int r = m0 + i * 16 + rowBase + v;
        int cc = n0 + j * 16 + colBase;
        if (r < M && cc < N) {
          float val = acc[i][j][v];
          if (FLAGS & GF_BIAS)  val += bias[cc];
          if (FLAGS & GF_GELU)  val = 0.5f * val * (1.0f + erff(val * 0.70710678118654752f));
          if (FLAGS & GF_RESID) val += resid[(long long)r * ldr + cc];
          long long o = coff + (long long)r * ldc + cc;
          if (FLAGS & GF_F32)  Cf[o] = val;
          if (FLAGS & GF_BF16) Cb[o] = __float2bfloat16(val);
        }
      }
}

// ---------------------------------------------------------------------------
// LayerNorm (cols = 768): one wave per row, f32 in -> bf16 out
// ---------------------------------------------------------------------------
__global__ __launch_bounds__(256)
void ln_kernel(const float* __restrict__ X, long long rowStride,
               const float* __restrict__ w, const float* __restrict__ b,
               __hip_bfloat16* __restrict__ Y, int ldy, int rows) {
  const int lane = threadIdx.x & 31;
  const int row = blockIdx.x * 8 + (threadIdx.x >> 5);
  if (row >= rows) return;
  const float* xr = X + row * rowStride;
  float vals[24], s = 0.f, s2 = 0.f;
#pragma unroll
  for (int i = 0; i < 24; ++i) {
    float v = xr[lane + 32 * i];
    vals[i] = v; s += v; s2 += v * v;
  }
#pragma unroll
  for (int o = 16; o >= 1; o >>= 1) {
    s  += __shfl_xor(s,  o, 32);
    s2 += __shfl_xor(s2, o, 32);
  }
  float mean = s * (1.f / 768.f);
  float var  = s2 * (1.f / 768.f) - mean * mean;
  float rstd = rsqrtf(var + 1e-5f);
  __hip_bfloat16* yr = Y + (long long)row * ldy;
#pragma unroll
  for (int i = 0; i < 24; ++i) {
    int c = lane + 32 * i;
    yr[c] = __float2bfloat16((vals[i] - mean) * rstd * w[c] + b[c]);
  }
}

// ---------------------------------------------------------------------------
// Softmax over score rows (197 valid of 224), scale = sqrt(64) = 8.
// Rewrites the f32 row IN PLACE as bf16 (wave32 lockstep: all loads precede
// stores). Pad cols 197..223 are written as bf16 zero.
// ---------------------------------------------------------------------------
__global__ __launch_bounds__(256)
void softmax_kernel(float* __restrict__ S, int rows) {
  const int lane = threadIdx.x & 31;
  const int row = blockIdx.x * 8 + (threadIdx.x >> 5);
  if (row >= rows) return;
  float* sr = S + (long long)row * SPAD;
  float v[7], mx = -3.0e38f;
#pragma unroll
  for (int i = 0; i < 7; ++i) {
    int c = lane + 32 * i;
    float x = (c < NT) ? sr[c] * 8.0f : -3.0e38f;
    v[i] = x; mx = fmaxf(mx, x);
  }
#pragma unroll
  for (int o = 16; o >= 1; o >>= 1) mx = fmaxf(mx, __shfl_xor(mx, o, 32));
  float sum = 0.f;
#pragma unroll
  for (int i = 0; i < 7; ++i) {
    int c = lane + 32 * i;
    v[i] = (c < NT) ? __expf(v[i] - mx) : 0.f;
    sum += v[i];
  }
#pragma unroll
  for (int o = 16; o >= 1; o >>= 1) sum += __shfl_xor(sum, o, 32);
  float inv = 1.f / sum;
  __hip_bfloat16* ar = reinterpret_cast<__hip_bfloat16*>(sr);
#pragma unroll
  for (int i = 0; i < 7; ++i) ar[lane + 32 * i] = __float2bfloat16(v[i] * inv);
}

// ---------------------------------------------------------------------------
// V transpose: Vt[bh][d][t] (t padded to 224 with zeros) from qkv bf16
// ---------------------------------------------------------------------------
__global__ void vt_kernel(const __hip_bfloat16* __restrict__ qkv,
                          __hip_bfloat16* __restrict__ Vt) {
  long long i = (long long)blockIdx.x * blockDim.x + threadIdx.x;
  const long long total = (long long)NBH * NHD * SPAD;
  if (i >= total) return;
  int t = (int)(i % SPAD);
  long long r = i / SPAD;
  int d = (int)(r % NHD);
  int bh = (int)(r / NHD);
  int b = bh / NH, h = bh % NH;
  __hip_bfloat16 val = __float2bfloat16(0.f);
  if (t < NT)
    val = qkv[((long long)(b * NT + t)) * (3 * ND) + 2 * ND + h * NHD + d];
  Vt[i] = val;
}

// ---------------------------------------------------------------------------
// Patch extraction: x[B,3,224,224] f32 -> patches bf16 [6272, 768]
// col = c*256 + py*16 + px  (matches conv-weight flatten (C, ph, pw))
// ---------------------------------------------------------------------------
__global__ void patches_kernel(const float* __restrict__ x,
                               __hip_bfloat16* __restrict__ out) {
  long long i = (long long)blockIdx.x * blockDim.x + threadIdx.x;
  const long long total = (long long)PROW * ND;
  if (i >= total) return;
  int col = (int)(i % ND);
  int row = (int)(i / ND);
  int b = row / NPAT, np = row % NPAT;
  int gy = np / 14, gx = np % 14;
  int c = col / 256, rem = col % 256, py = rem / 16, px = rem % 16;
  float v = x[(((long long)b * 3 + c) * 224 + (gy * 16 + py)) * 224 + (gx * 16 + px)];
  out[i] = __float2bfloat16(v);
}

// ---------------------------------------------------------------------------
// Assemble token stream: h[b*197 + 0] = cls; h[b*197+1+np] = patchOut + pos
// ---------------------------------------------------------------------------
__global__ void build_h_kernel(const float* __restrict__ pe,
                               const float* __restrict__ pos,
                               const float* __restrict__ cls,
                               float* __restrict__ h) {
  long long i = (long long)blockIdx.x * blockDim.x + threadIdx.x;
  const long long total = (long long)TOK * ND;
  if (i >= total) return;
  int d = (int)(i % ND);
  int row = (int)(i / ND);
  int b = row / NT, t = row % NT;
  float v;
  if (t == 0) v = cls[d];
  else        v = pe[((long long)b * NPAT + (t - 1)) * ND + d] + pos[(long long)(t - 1) * ND + d];
  h[i] = v;
}

__global__ void f32_to_bf16_kernel(const float* __restrict__ src,
                                   __hip_bfloat16* __restrict__ dst, long long n) {
  long long i = (long long)blockIdx.x * blockDim.x + threadIdx.x;
  if (i < n) dst[i] = __float2bfloat16(src[i]);
}

// ---------------------------------------------------------------------------
// Host-side helpers
// ---------------------------------------------------------------------------
template <int F>
static inline void gemm_go(hipStream_t s,
                           const __hip_bfloat16* A, int lda,
                           const __hip_bfloat16* W, int ldw,
                           float* Cf, __hip_bfloat16* Cb, int ldc,
                           const float* bias, const float* resid, int ldr,
                           int M, int N, int K, int Z = 1, int Hdim = 1,
                           long long saB = 0, long long saH = 0,
                           long long swB = 0, long long swH = 0,
                           long long scB = 0, long long scH = 0) {
  dim3 grid((N + 63) / 64, (M + 127) / 128, Z);
  gemm_wmma<F><<<grid, 256, 0, s>>>(A, lda, W, ldw, Cf, Cb, ldc, bias, resid,
                                    ldr, M, N, K, saB, saH, swB, swH, scB, scH, Hdim);
}

static inline void conv_go(hipStream_t s, const float* src, __hip_bfloat16* dst, long long n) {
  f32_to_bf16_kernel<<<(unsigned)((n + 255) / 256), 256, 0, s>>>(src, dst, n);
}

extern "C" void kernel_launch(void* const* d_in, const int* in_sizes, int n_in,
                              void* d_out, int out_size, void* d_ws, size_t ws_size,
                              hipStream_t stream) {
  (void)in_sizes; (void)n_in; (void)out_size; (void)ws_size;

  const float* x        = (const float*)d_in[0];
  const float* patch_w  = (const float*)d_in[1];
  const float* patch_b  = (const float*)d_in[2];
  const float* pos      = (const float*)d_in[3];
  const float* cls      = (const float*)d_in[4];
  const float* ln1_w    = (const float*)d_in[5];
  const float* ln1_b    = (const float*)d_in[6];
  const float* qkv_w    = (const float*)d_in[7];
  const float* qkv_b    = (const float*)d_in[8];
  const float* proj_w   = (const float*)d_in[9];
  const float* proj_b   = (const float*)d_in[10];
  const float* ln2_w    = (const float*)d_in[11];
  const float* ln2_b    = (const float*)d_in[12];
  const float* fc1_w    = (const float*)d_in[13];
  const float* fc1_b    = (const float*)d_in[14];
  const float* fc2_w    = (const float*)d_in[15];
  const float* fc2_b    = (const float*)d_in[16];
  const float* lnf_w    = (const float*)d_in[17];
  const float* lnf_b    = (const float*)d_in[18];
  const float* head_w   = (const float*)d_in[19];
  const float* head_b   = (const float*)d_in[20];

  // ----- workspace carve-up -----
  char* ws = (char*)d_ws;
  size_t off = 0;
  auto carve = [&](size_t bytes) {
    size_t o = off;
    off += (bytes + 255) & ~(size_t)255;
    return (void*)(ws + o);
  };
  typedef __hip_bfloat16 bf;
  bf* wPatch = (bf*)carve((size_t)ND * ND * 2);
  bf* wQkv   = (bf*)carve((size_t)NL * 3 * ND * ND * 2);
  bf* wProj  = (bf*)carve((size_t)NL * ND * ND * 2);
  bf* wFc1   = (bf*)carve((size_t)NL * 4 * ND * ND * 2);
  bf* wFc2   = (bf*)carve((size_t)NL * 4 * ND * ND * 2);
  bf* wHead  = (bf*)carve((size_t)NC * ND * 2);
  float* hbuf = (float*)carve((size_t)TOK * ND * 4);
  bf* s1     = (bf*)carve((size_t)TOK * ND * 2);            // patches / LN outputs
  bf* qkvb   = (bf*)carve((size_t)(TOK + 32) * 3 * ND * 2); // qkv bf16
  bf* vtb    = (bf*)carve((size_t)NBH * NHD * SPAD * 2);    // V transposed
  bf* ctxb   = (bf*)carve((size_t)TOK * ND * 2);            // attention context
  bf* clsb   = (bf*)carve((size_t)NB * ND * 2);             // final cls rows
  // big aliased scratch: patch-embed f32 temp | scores f32 (att bf16 in-place) | MLP hidden bf16
  size_t bigBytes = (size_t)NBH * NT * SPAD * 4;            // 67.8 MB (largest user)
  void* big = carve(bigBytes);
  float* petmp  = (float*)big;
  float* scores = (float*)big;
  bf*    attb   = (bf*)big;        // same bytes, bf16 view (row stride 448 elems)
  bf*    mlpb   = (bf*)big;

  // ----- weight conversion (deterministic, every call) -----
  conv_go(stream, patch_w, wPatch, (long long)ND * ND);
  conv_go(stream, qkv_w,   wQkv,   (long long)NL * 3 * ND * ND);
  conv_go(stream, proj_w,  wProj,  (long long)NL * ND * ND);
  conv_go(stream, fc1_w,   wFc1,   (long long)NL * 4 * ND * ND);
  conv_go(stream, fc2_w,   wFc2,   (long long)NL * 4 * ND * ND);
  conv_go(stream, head_w,  wHead,  (long long)NC * ND);

  // ----- patch embedding -----
  {
    long long n = (long long)PROW * ND;
    patches_kernel<<<(unsigned)((n + 255) / 256), 256, 0, stream>>>(x, s1);
  }
  gemm_go<GF_BIAS | GF_F32>(stream, s1, ND, wPatch, ND, petmp, nullptr, ND,
                            patch_b, nullptr, 0, PROW, ND, ND);
  {
    long long n = (long long)TOK * ND;
    build_h_kernel<<<(unsigned)((n + 255) / 256), 256, 0, stream>>>(petmp, pos, cls, hbuf);
  }

  // ----- transformer layers -----
  const int lnGrid = (TOK + 7) / 8;
  for (int l = 0; l < NL; ++l) {
    // LN1 -> s1 (bf16)
    ln_kernel<<<lnGrid, 256, 0, stream>>>(hbuf, ND, ln1_w + l * ND, ln1_b + l * ND,
                                          s1, ND, TOK);
    // QKV: [6304,768] x [2304,768]^T -> bf16 qkvb
    gemm_go<GF_BIAS | GF_BF16>(stream, s1, ND, wQkv + (size_t)l * 3 * ND * ND, ND,
                               nullptr, qkvb, 3 * ND, qkv_b + l * 3 * ND, nullptr, 0,
                               TOK, 3 * ND, ND);
    // V transpose
    {
      long long n = (long long)NBH * NHD * SPAD;
      vt_kernel<<<(unsigned)((n + 255) / 256), 256, 0, stream>>>(qkvb, vtb);
    }
    // scores = Q K^T  (batched over 384 (b,h)) -> f32 [bh][197][224]
    gemm_go<GF_F32>(stream, qkvb, 3 * ND, qkvb + ND, 3 * ND, scores, nullptr, SPAD,
                    nullptr, nullptr, 0, NT, NT, NHD, NBH, NH,
                    (long long)NT * 3 * ND, NHD,
                    (long long)NT * 3 * ND, NHD,
                    (long long)NH * NT * SPAD, (long long)NT * SPAD);
    // softmax (x8 scale) in place -> bf16 att rows (stride 448 bf16)
    softmax_kernel<<<(NBH * NT + 7) / 8, 256, 0, stream>>>(scores, NBH * NT);
    // O = att @ V  -> ctx bf16 [6304,768]
    gemm_go<GF_BF16>(stream, attb, 2 * SPAD, vtb, SPAD, nullptr, ctxb, ND,
                     nullptr, nullptr, 0, NT, NHD, SPAD, NBH, NH,
                     (long long)NH * NT * 2 * SPAD, (long long)NT * 2 * SPAD,
                     (long long)NH * NHD * SPAD,   (long long)NHD * SPAD,
                     (long long)NT * ND, NHD);
    // proj + residual -> h (f32)
    gemm_go<GF_BIAS | GF_RESID | GF_F32>(stream, ctxb, ND,
                                         wProj + (size_t)l * ND * ND, ND,
                                         hbuf, nullptr, ND, proj_b + l * ND,
                                         hbuf, ND, TOK, ND, ND);
    // LN2 -> s1
    ln_kernel<<<lnGrid, 256, 0, stream>>>(hbuf, ND, ln2_w + l * ND, ln2_b + l * ND,
                                          s1, ND, TOK);
    // FC1 + GELU -> mlpb bf16 [6304,3072]
    gemm_go<GF_BIAS | GF_GELU | GF_BF16>(stream, s1, ND,
                                         wFc1 + (size_t)l * 4 * ND * ND, ND,
                                         nullptr, mlpb, 4 * ND, fc1_b + l * 4 * ND,
                                         nullptr, 0, TOK, 4 * ND, ND);
    // FC2 + residual -> h
    gemm_go<GF_BIAS | GF_RESID | GF_F32>(stream, mlpb, 4 * ND,
                                         wFc2 + (size_t)l * 4 * ND * ND, 4 * ND,
                                         hbuf, nullptr, ND, fc2_b + l * ND,
                                         hbuf, ND, TOK, ND, 4 * ND);
  }

  // ----- final LN on cls rows + head -----
  ln_kernel<<<(NB + 7) / 8, 256, 0, stream>>>(hbuf, (long long)NT * ND,
                                              lnf_w, lnf_b, clsb, ND, NB);
  gemm_go<GF_BIAS | GF_F32>(stream, clsb, ND, wHead, ND, (float*)d_out, nullptr,
                            NC, head_b, nullptr, 0, NB, NC, ND);
}
